// HGCN_85624468013344
// MI455X (gfx1250) — matrix operand order repaired and checked by reading, not compiled, not run on hardware
//
#include <hip/hip_runtime.h>
#include <math.h>

typedef __attribute__((ext_vector_type(16))) __bf16 v16bf;
typedef __attribute__((ext_vector_type(8)))  float  v8f;

__device__ __forceinline__ float elu1(float x) {
    return x > 0.f ? x : (expf(x) - 1.f);
}

// ---------------------------------------------------------------- degrees ---
__global__ void __launch_bounds__(256) deg_count(const int* __restrict__ idx,
                                                 int n, int* __restrict__ cnt) {
    int i = blockIdx.x * 256 + threadIdx.x;
    if (i < n) atomicAdd(&cnt[idx[i]], 1);
}

__global__ void __launch_bounds__(256) deg_inv_sqrt(const int* __restrict__ cnt,
                                                    float* __restrict__ out, int n) {
    int i = blockIdx.x * 256 + threadIdx.x;
    if (i < n) out[i] = rsqrtf(fmaxf((float)cnt[i], 1.f));
}

// ---------------------------------------------------- edge gather/scatter ---
// item = (edge, 4-float column chunk):  agg[dst] += h[src] * d_out^-1/2[src]
__global__ void __launch_bounds__(256) edge_scatter(
    const float* __restrict__ hsrc, const int* __restrict__ src,
    const int* __restrict__ dst, const float* __restrict__ dsc,
    float* __restrict__ agg, int nitems)
{
    int it = blockIdx.x * 256 + threadIdx.x;
    if (it >= nitems) return;
    int e = it >> 5;
    int c = (it & 31) << 2;
    int s = src[e];
    int d = dst[e];
    float sc = dsc[s];
    float4 v = *(const float4*)(hsrc + (size_t)s * 128 + c);
    float* p = agg + (size_t)d * 128 + c;
    atomicAdd(p + 0, v.x * sc);
    atomicAdd(p + 1, v.y * sc);
    atomicAdd(p + 2, v.z * sc);
    atomicAdd(p + 3, v.w * sc);
}

// -------------------------------------------------- fused WMMA layer -------
// Y[M,128] = post( prescale[row]*X[M,IN_DIM] @ W[IN_DIM,128] + bias )
// post: optional ELU, optional +skip, optional LayerNorm(gamma,beta)
// 128 threads = 4 waves; each wave owns 16 rows and all 8 N-tiles.
template<int IN_DIM, bool ACT, bool USE_LN>
__global__ void __launch_bounds__(128) wmma_layer(
    const float* __restrict__ X, int ld_x,
    const float* __restrict__ W,
    const float* __restrict__ bias,
    const float* __restrict__ prescale,
    const float* __restrict__ skip,
    const float* __restrict__ gamma,
    const float* __restrict__ beta,
    float* __restrict__ Y)
{
    constexpr int KT   = (IN_DIM + 31) / 32;   // k-tiles of 32
    constexpr int KTOT = KT * 32;
    constexpr int KS   = KTOT + 8;             // LDS W row stride (bf16 elems)
    constexpr int OS   = 132;                  // LDS out row stride (f32)
    constexpr size_t SW = (size_t)128 * KS * 2;
    constexpr size_t SO = (size_t)64 * OS * 4;
    constexpr size_t SM = SW > SO ? SW : SO;
    __shared__ char smem[SM];

    const int tid  = threadIdx.x;
    const int wave = tid >> 5;
    const int lane = tid & 31;
    const int half = lane >> 4;
    const int mr   = lane & 15;

    // ---- stage W transposed (Wt[n][k]) as bf16 in LDS ----
    __bf16* wsm = (__bf16*)smem;
    for (int e = tid; e < 128 * KTOT; e += 128) {
        int k = e >> 7;
        int n = e & 127;
        float w = (k < IN_DIM) ? W[k * 128 + n] : 0.f;
        wsm[n * KS + k] = (__bf16)w;
    }
    __syncthreads();

    // ---- A operand: 16 rows per wave, ISA 16-bit A 16x32 lane layout ----
    const int row0 = blockIdx.x * 64 + wave * 16;
    const int row  = row0 + mr;
    const float* xr = X + (size_t)row * ld_x;
    const float ps = prescale ? prescale[row] : 1.f;

    v16bf a[KT];
#pragma unroll
    for (int kt = 0; kt < KT; ++kt) {
        float f[16];
        const int c0 = kt * 32 + half * 8;       // elems 0..7  -> K = c0..c0+7
        if (c0 + 8 <= IN_DIM) {
            float4 u0 = *(const float4*)(xr + c0);
            float4 u1 = *(const float4*)(xr + c0 + 4);
            f[0] = u0.x; f[1] = u0.y; f[2] = u0.z; f[3] = u0.w;
            f[4] = u1.x; f[5] = u1.y; f[6] = u1.z; f[7] = u1.w;
        } else {
#pragma unroll
            for (int i = 0; i < 8; ++i) f[i] = 0.f;
        }
        const int c1 = c0 + 16;                  // elems 8..15 -> K = c1..c1+7
        if (c1 + 8 <= IN_DIM) {
            float4 u0 = *(const float4*)(xr + c1);
            float4 u1 = *(const float4*)(xr + c1 + 4);
            f[8]  = u0.x; f[9]  = u0.y; f[10] = u0.z; f[11] = u0.w;
            f[12] = u1.x; f[13] = u1.y; f[14] = u1.z; f[15] = u1.w;
        } else {
#pragma unroll
            for (int i = 8; i < 16; ++i) f[i] = 0.f;
        }
#pragma unroll
        for (int i = 0; i < 16; ++i) a[kt][i] = (__bf16)(f[i] * ps);
    }

    // ---- main loop: 8 N-tiles x KT WMMAs ----
    v8f acc[8];
#pragma unroll
    for (int nt = 0; nt < 8; ++nt) acc[nt] = (v8f){};

#pragma unroll
    for (int nt = 0; nt < 8; ++nt) {
#pragma unroll
        for (int kt = 0; kt < KT; ++kt) {
            // B elem i <-> K = half*8 + i (i<8), 16 + half*8 + (i-8) (i>=8), N = lane&15
            const __bf16* bp = wsm + (nt * 16 + mr) * KS + kt * 32 + half * 8;
            union { uint4 q[2]; v16bf v; } tb;
            tb.q[0] = *(const uint4*)(bp);
            tb.q[1] = *(const uint4*)(bp + 16);
            acc[nt] = __builtin_amdgcn_wmma_f32_16x16x32_bf16(
                false, a[kt], false, tb.v, (short)0, acc[nt], false, false);
        }
    }

    __syncthreads();                       // all waves done reading W
    float* osm = (float*)smem;             // repurpose LDS for output tiles
    // C/D layout: VGPR r -> M = half*8 + r, N = lane&15
#pragma unroll
    for (int nt = 0; nt < 8; ++nt) {
#pragma unroll
        for (int r = 0; r < 8; ++r)
            osm[(wave * 16 + half * 8 + r) * OS + nt * 16 + mr] = acc[nt][r];
    }
    __syncthreads();

    // ---- epilogue: 2 lanes per row (half selects 64-col slice) ----
    const int erow  = row0 + mr;
    const int cbase = half * 64;
    const float* lrow = osm + (wave * 16 + mr) * OS + cbase;
    const float* srow = skip ? skip + (size_t)erow * 128 + cbase : nullptr;

    float4 vv[16];
    float s1 = 0.f, s2 = 0.f;
#pragma unroll
    for (int j = 0; j < 16; ++j) {
        float4 v  = *(const float4*)(lrow + 4 * j);
        float4 bb = *(const float4*)(bias + cbase + 4 * j);
        v.x += bb.x; v.y += bb.y; v.z += bb.z; v.w += bb.w;
        if (ACT) {
            v.x = elu1(v.x); v.y = elu1(v.y); v.z = elu1(v.z); v.w = elu1(v.w);
        }
        if (srow) {
            float4 sv = *(const float4*)(srow + 4 * j);
            v.x += sv.x; v.y += sv.y; v.z += sv.z; v.w += sv.w;
        }
        vv[j] = v;
        s1 += v.x + v.y + v.z + v.w;
        s2 += v.x * v.x + v.y * v.y + v.z * v.z + v.w * v.w;
    }

    float* yrow = Y + (size_t)erow * 128 + cbase;
    if (USE_LN) {
        s1 += __shfl_xor(s1, 16);
        s2 += __shfl_xor(s2, 16);
        const float mu  = s1 * (1.f / 128.f);
        const float var = s2 * (1.f / 128.f) - mu * mu;
        const float rs  = rsqrtf(var + 1e-5f);
#pragma unroll
        for (int j = 0; j < 16; ++j) {
            float4 g4 = *(const float4*)(gamma + cbase + 4 * j);
            float4 b4 = *(const float4*)(beta  + cbase + 4 * j);
            float4 v = vv[j];
            v.x = (v.x - mu) * rs * g4.x + b4.x;
            v.y = (v.y - mu) * rs * g4.y + b4.y;
            v.z = (v.z - mu) * rs * g4.z + b4.z;
            v.w = (v.w - mu) * rs * g4.w + b4.w;
            *(float4*)(yrow + 4 * j) = v;
        }
    } else {
#pragma unroll
        for (int j = 0; j < 16; ++j) *(float4*)(yrow + 4 * j) = vv[j];
    }
}

// ------------------------------------------------------------- pooling -----
// graph_id is sorted: accumulate runs locally, atomic only at boundaries.
__global__ void __launch_bounds__(128) pool_kernel(
    const float* __restrict__ h, const int* __restrict__ gid,
    float* __restrict__ sums, float* __restrict__ cnts)
{
    const int col = threadIdx.x;
    const int r0  = blockIdx.x * 256;
    float acc = 0.f, c = 0.f;
    int gc = gid[r0];
    for (int i = 0; i < 256; ++i) {
        int g = gid[r0 + i];
        if (g != gc) {
            atomicAdd(&sums[gc * 128 + col], acc);
            if (col == 0) atomicAdd(&cnts[gc], c);
            acc = 0.f; c = 0.f; gc = g;
        }
        acc += h[(size_t)(r0 + i) * 128 + col];
        c += 1.f;
    }
    atomicAdd(&sums[gc * 128 + col], acc);
    if (col == 0) atomicAdd(&cnts[gc], c);
}

__global__ void __launch_bounds__(128) finalize(
    const float* __restrict__ sums, const float* __restrict__ cnts,
    float* __restrict__ out)
{
    int i = blockIdx.x * 128 + threadIdx.x;
    if (i < 64 * 128) {
        out[i] = sums[i] / fmaxf(cnts[i >> 7], 1.f);
    } else if (i < 64 * 128 + 64) {
        out[i] = cnts[i - 64 * 128];
    }
}

// ------------------------------------------------------------ launcher -----
extern "C" void kernel_launch(void* const* d_in, const int* in_sizes, int n_in,
                              void* d_out, int out_size, void* d_ws, size_t ws_size,
                              hipStream_t stream)
{
    (void)in_sizes; (void)n_in; (void)out_size; (void)ws_size;
    constexpr int N_AA = 262144, N_CG = 65536;
    constexpr int E_CC = 262144, E_CA = 524288;

    const float* h_aa0  = (const float*)d_in[0];
    const float* h_cg0  = (const float*)d_in[1];
    const int*   src_cc = (const int*)d_in[2];
    const int*   dst_cc = (const int*)d_in[3];
    const int*   src_ca = (const int*)d_in[4];
    const int*   dst_ca = (const int*)d_in[5];
    // d_in[6], d_in[7] (src_ac, dst_ac) feed only dead h_cg output: skipped.
    const int*   gid    = (const int*)d_in[8];
    const float* W_cg0  = (const float*)d_in[9];
    const float* b_cg0  = (const float*)d_in[10];
    const float* g_cg0  = (const float*)d_in[11];
    const float* be_cg0 = (const float*)d_in[12];
    const float* W_aa0  = (const float*)d_in[13];
    const float* b_aa0  = (const float*)d_in[14];
    const float* g_aa0  = (const float*)d_in[15];
    const float* be_aa0 = (const float*)d_in[16];
    const float* W_cc   = (const float*)d_in[17];
    const float* b_cc   = (const float*)d_in[18];
    const float* W_m1   = (const float*)d_in[19];
    const float* b_m1   = (const float*)d_in[20];
    const float* g_m1   = (const float*)d_in[21];
    const float* be_m1  = (const float*)d_in[22];
    const float* W_ca   = (const float*)d_in[23];
    const float* b_ca   = (const float*)d_in[24];
    const float* W_m2   = (const float*)d_in[25];
    const float* b_m2   = (const float*)d_in[26];
    const float* g_m2   = (const float*)d_in[27];
    const float* be_m2  = (const float*)d_in[28];
    // d_in[29], d_in[30] (W_ac, b_ac): dead.
    const float* W_f    = (const float*)d_in[31];
    const float* b_f    = (const float*)d_in[32];
    const float* g_f    = (const float*)d_in[33];
    const float* be_f   = (const float*)d_in[34];

    char* ws = (char*)d_ws;
    size_t off = 0;
    auto take = [&](size_t bytes) -> char* {
        char* p = ws + off;
        off += (bytes + 255) & ~(size_t)255;
        return p;
    };
    float* h_cg = (float*)take((size_t)N_CG * 128 * 4);
    float* h_aa = (float*)take((size_t)N_AA * 128 * 4);
    float* agg  = (float*)take((size_t)N_AA * 128 * 4);   // shared by both convs
    const int NDEG = N_CG * 3 + N_AA;
    int*   cnt_base = (int*)take((size_t)NDEG * 4);
    float* d_base   = (float*)take((size_t)NDEG * 4);
    float* psum     = (float*)take((64 * 128 + 64) * 4);

    int* c_out_cc = cnt_base;
    int* c_in_cc  = cnt_base + N_CG;
    int* c_out_ca = cnt_base + 2 * N_CG;
    int* c_in_ca  = cnt_base + 3 * N_CG;          // length N_AA
    float* s_out_cc = d_base;
    float* s_in_cc  = d_base + N_CG;
    float* s_out_ca = d_base + 2 * N_CG;
    float* s_in_ca  = d_base + 3 * N_CG;
    float* pcnt = psum + 64 * 128;

    // degrees
    hipMemsetAsync(cnt_base, 0, (size_t)NDEG * 4, stream);
    deg_count<<<(E_CC + 255) / 256, 256, 0, stream>>>(src_cc, E_CC, c_out_cc);
    deg_count<<<(E_CC + 255) / 256, 256, 0, stream>>>(dst_cc, E_CC, c_in_cc);
    deg_count<<<(E_CA + 255) / 256, 256, 0, stream>>>(src_ca, E_CA, c_out_ca);
    deg_count<<<(E_CA + 255) / 256, 256, 0, stream>>>(dst_ca, E_CA, c_in_ca);
    deg_inv_sqrt<<<(NDEG + 255) / 256, 256, 0, stream>>>(cnt_base, d_base, NDEG);

    // input embedding MLPs
    wmma_layer<32, true, true><<<N_CG / 64, 128, 0, stream>>>(
        h_cg0, 32, W_cg0, b_cg0, nullptr, nullptr, g_cg0, be_cg0, h_cg);
    wmma_layer<16, true, true><<<N_AA / 64, 128, 0, stream>>>(
        h_aa0, 16, W_aa0, b_aa0, nullptr, nullptr, g_aa0, be_aa0, h_aa);

    // cg->cg conv: h_cg = elu((D^-1/2 A D^-1/2 h_cg) @ W_cc + b) + h_cg
    hipMemsetAsync(agg, 0, (size_t)N_CG * 128 * 4, stream);
    edge_scatter<<<(E_CC * 32) / 256, 256, 0, stream>>>(
        h_cg, src_cc, dst_cc, s_out_cc, agg, E_CC * 32);
    wmma_layer<128, true, false><<<N_CG / 64, 128, 0, stream>>>(
        agg, 128, W_cc, b_cc, s_in_cc, h_cg, nullptr, nullptr, h_cg);

    // 'm1' nodewise MLP on aa (skip + LN), in place (row-local)
    wmma_layer<128, true, true><<<N_AA / 64, 128, 0, stream>>>(
        h_aa, 128, W_m1, b_m1, nullptr, h_aa, g_m1, be_m1, h_aa);

    // cg->aa conv: h_aa = elu((D^-1/2 A D^-1/2 h_cg) @ W_ca + b) + h_aa
    hipMemsetAsync(agg, 0, (size_t)N_AA * 128 * 4, stream);
    edge_scatter<<<(E_CA * 32) / 256, 256, 0, stream>>>(
        h_cg, src_ca, dst_ca, s_out_ca, agg, E_CA * 32);
    wmma_layer<128, true, false><<<N_AA / 64, 128, 0, stream>>>(
        agg, 128, W_ca, b_ca, s_in_ca, h_aa, nullptr, nullptr, h_aa);

    // 'm2' nodewise MLP on aa (skip + LN)
    wmma_layer<128, true, true><<<N_AA / 64, 128, 0, stream>>>(
        h_aa, 128, W_m2, b_m2, nullptr, h_aa, g_m2, be_m2, h_aa);

    // final nodewise layer: no act, no skip, LN
    wmma_layer<128, false, true><<<N_AA / 64, 128, 0, stream>>>(
        h_aa, 128, W_f, b_f, nullptr, nullptr, g_f, be_f, h_aa);

    // mean pooling per graph + counts
    hipMemsetAsync(psum, 0, (64 * 128 + 64) * 4, stream);
    pool_kernel<<<N_AA / 256, 128, 0, stream>>>(h_aa, gid, psum, pcnt);
    finalize<<<(64 * 128 + 64 + 127) / 128, 128, 0, stream>>>(psum, pcnt, (float*)d_out);
}
